// GRU_31903017075235
// MI455X (gfx1250) — matrix operand order repaired and checked by reading, not compiled, not run on hardware
//
#include <hip/hip_runtime.h>
#include <hip/hip_bf16.h>

typedef __bf16 bf16;
typedef __attribute__((ext_vector_type(16))) __bf16 v16bf;
typedef __attribute__((ext_vector_type(8)))  float  v8f;

#define B_   256
#define C_   2048
#define H_   1024
#define T_   64
#define G3H  (3 * H_)

// LDS B-tile: fragment order [nFrag(8)][lane(32)][16 bf16], double buffered.
#define BUF_ELEMS (8 * 32 * 16)

__device__ __forceinline__ float sigmoidf_(float x) {
    return 1.0f / (1.0f + __expf(-x));
}

// ---------------------------------------------------------------------------
// WMMA A fragment: 16(M) x 32(K) bf16 from row-major X (leading dim ldx).
//   lanes 0-15: M=lane,   VGPR0-3 = K 0..7,  VGPR4-7 = K 16..23
//   lanes16-31: M=lane-16,VGPR0-3 = K 8..15, VGPR4-7 = K 24..31
// ---------------------------------------------------------------------------
__device__ __forceinline__ v16bf load_a_frag(const bf16* X, int ldx, int m0, int k0) {
    const int lane = threadIdx.x & 31;
    const int m    = m0 + (lane & 15);
    const int kb   = k0 + ((lane >> 4) << 3);       // +0 or +8
    const bf16* p  = X + (size_t)m * ldx + kb;
    union { uint4 u[2]; v16bf v; } t;
    t.u[0] = *(const uint4*)(p);                    // K kb..kb+7
    t.u[1] = *(const uint4*)(p + 16);               // K kb+16..kb+23
    return t.v;
}

// ===========================================================================
// Kernel: fused gi / gh gate GEMMs. blockIdx.z: 0 -> gi, 1 -> gh.
// Block = 256 threads = 8 waves, 4(M) x 2(N) -> 64 x 128 block tile.
// Weight tile staged through double-buffered LDS (shared by the 4 M-waves).
// ===========================================================================
__global__ __launch_bounds__(256) void gates_gemm_kernel(
    const bf16* __restrict__ codes_bf,  // [B, C]
    const bf16* __restrict__ h_bf,      // [B, H]
    const bf16* __restrict__ Wih_t,     // [C, 3H]  (K-major)
    const bf16* __restrict__ Whh_t,     // [H, 3H]  (K-major)
    const float* __restrict__ b_ih,     // [3H]
    const float* __restrict__ b_hh,     // [3H]
    float* __restrict__ gi,             // [B, 3H]
    float* __restrict__ gh)             // [B, 3H]
{
    __shared__ __align__(16) bf16 Bsh[2][8][32][16];

    const int tid  = threadIdx.x;
    const int lane = tid & 31;
    const int wave = tid >> 5;
    const int m0   = blockIdx.y * 64 + (wave & 3) * 16;
    const int wn   = wave >> 2;                       // 0..1
    const int n0   = blockIdx.x * 128 + wn * 64;
    const int nblk = blockIdx.x * 128;

    const bf16*  X;  const bf16* Wt; const float* bias; float* O; int K, ldx;
    if (blockIdx.z == 0) { X = codes_bf; ldx = C_; Wt = Wih_t; bias = b_ih; O = gi; K = C_; }
    else                 { X = h_bf;     ldx = H_; Wt = Whh_t; bias = b_hh; O = gh; K = H_; }

    // Staging map: chunk c in [0,512) -> k-row = c>>4, u = c&15 covers n bytes
    // [u*16, u*16+16) of the 256-byte tile row. Thread handles chunks tid, tid+256.
    const int k_s0 = tid >> 4,        u0 = tid & 15;
    const int k_s1 = (tid >> 4) + 16, u1 = tid & 15;
    const bf16* src0 = Wt + (size_t)k_s0 * G3H + nblk + u0 * 8;
    const bf16* src1 = Wt + (size_t)k_s1 * G3H + nblk + u1 * 8;
    bf16* dst0 = &Bsh[0][u0 >> 1][(u0 & 1) * 8 + ((k_s0 >> 3) & 1) * 16][(k_s0 & 7) + ((k_s0 >> 4) << 3)];
    bf16* dst1 = &Bsh[0][u1 >> 1][(u1 & 1) * 8 + ((k_s1 >> 3) & 1) * 16][(k_s1 & 7) + ((k_s1 >> 4) << 3)];

    v8f acc[4];
    acc[0] = v8f{}; acc[1] = v8f{}; acc[2] = v8f{}; acc[3] = v8f{};

    uint4 r0 = *(const uint4*)(src0);
    uint4 r1 = *(const uint4*)(src1);
    v16bf a  = load_a_frag(X, ldx, m0, 0);
    int buf = 0;
    for (int k = 0; k < K; k += 32) {
        for (int j = 0; j < 8; ++j) {
            dst0[buf * BUF_ELEMS + j * 16] = ((const bf16*)&r0)[j];
            dst1[buf * BUF_ELEMS + j * 16] = ((const bf16*)&r1)[j];
        }
        __syncthreads();
        v16bf a_next;
        if (k + 32 < K) {
            r0     = *(const uint4*)(src0 + (size_t)(k + 32) * G3H);
            r1     = *(const uint4*)(src1 + (size_t)(k + 32) * G3H);
            a_next = load_a_frag(X, ldx, m0, k + 32);
        }
        // Load all four B fragments first so the ds_loads pipeline.
        union { uint4 u[2]; v16bf v; } tb0, tb1, tb2, tb3;
        {
            const bf16* p0 = &Bsh[buf][wn * 4 + 0][lane][0];
            const bf16* p1 = &Bsh[buf][wn * 4 + 1][lane][0];
            const bf16* p2 = &Bsh[buf][wn * 4 + 2][lane][0];
            const bf16* p3 = &Bsh[buf][wn * 4 + 3][lane][0];
            tb0.u[0] = *(const uint4*)(p0); tb0.u[1] = *(const uint4*)(p0 + 8);
            tb1.u[0] = *(const uint4*)(p1); tb1.u[1] = *(const uint4*)(p1 + 8);
            tb2.u[0] = *(const uint4*)(p2); tb2.u[1] = *(const uint4*)(p2 + 8);
            tb3.u[0] = *(const uint4*)(p3); tb3.u[1] = *(const uint4*)(p3 + 8);
        }
        acc[0] = __builtin_amdgcn_wmma_f32_16x16x32_bf16(false, a, false, tb0.v, (short)0, acc[0], false, false);
        acc[1] = __builtin_amdgcn_wmma_f32_16x16x32_bf16(false, a, false, tb1.v, (short)0, acc[1], false, false);
        acc[2] = __builtin_amdgcn_wmma_f32_16x16x32_bf16(false, a, false, tb2.v, (short)0, acc[2], false, false);
        acc[3] = __builtin_amdgcn_wmma_f32_16x16x32_bf16(false, a, false, tb3.v, (short)0, acc[3], false, false);
        a = a_next;
        buf ^= 1;
    }

    const int row = m0 + ((lane >> 4) << 3);
#pragma unroll
    for (int i = 0; i < 4; ++i) {
        const int col = n0 + 16 * i + (lane & 15);
        const float bv = bias[col];
#pragma unroll
        for (int r = 0; r < 8; ++r)
            O[(size_t)(row + r) * G3H + col] = acc[i][r] + bv;
    }
}

// ===========================================================================
// Kernel: logits = X @ Wout^T + b_out, sigmoid; write samples slice (f32)
// and next-step codes (bf16). M=256, N=2048, K=1024. Same LDS staging.
// ===========================================================================
__global__ __launch_bounds__(256) void out_sigmoid_gemm_kernel(
    const bf16* __restrict__ Xbf,       // [B, H]
    const bf16* __restrict__ Wout_t,    // [H, C]  (K-major)
    const float* __restrict__ b_out,    // [C]
    float* __restrict__ samples_t,      // samples + t*C ; row stride T*C
    bf16* __restrict__ codes_bf)        // [B, C]
{
    __shared__ __align__(16) bf16 Bsh[2][8][32][16];

    const int tid  = threadIdx.x;
    const int lane = tid & 31;
    const int wave = tid >> 5;
    const int m0   = blockIdx.y * 64 + (wave & 3) * 16;
    const int wn   = wave >> 2;
    const int n0   = blockIdx.x * 128 + wn * 64;
    const int nblk = blockIdx.x * 128;

    const int k_s0 = tid >> 4,        u0 = tid & 15;
    const int k_s1 = (tid >> 4) + 16, u1 = tid & 15;
    const bf16* src0 = Wout_t + (size_t)k_s0 * C_ + nblk + u0 * 8;
    const bf16* src1 = Wout_t + (size_t)k_s1 * C_ + nblk + u1 * 8;
    bf16* dst0 = &Bsh[0][u0 >> 1][(u0 & 1) * 8 + ((k_s0 >> 3) & 1) * 16][(k_s0 & 7) + ((k_s0 >> 4) << 3)];
    bf16* dst1 = &Bsh[0][u1 >> 1][(u1 & 1) * 8 + ((k_s1 >> 3) & 1) * 16][(k_s1 & 7) + ((k_s1 >> 4) << 3)];

    v8f acc[4];
    acc[0] = v8f{}; acc[1] = v8f{}; acc[2] = v8f{}; acc[3] = v8f{};

    uint4 r0 = *(const uint4*)(src0);
    uint4 r1 = *(const uint4*)(src1);
    v16bf a  = load_a_frag(Xbf, H_, m0, 0);
    int buf = 0;
    for (int k = 0; k < H_; k += 32) {
        for (int j = 0; j < 8; ++j) {
            dst0[buf * BUF_ELEMS + j * 16] = ((const bf16*)&r0)[j];
            dst1[buf * BUF_ELEMS + j * 16] = ((const bf16*)&r1)[j];
        }
        __syncthreads();
        v16bf a_next;
        if (k + 32 < H_) {
            r0     = *(const uint4*)(src0 + (size_t)(k + 32) * C_);
            r1     = *(const uint4*)(src1 + (size_t)(k + 32) * C_);
            a_next = load_a_frag(Xbf, H_, m0, k + 32);
        }
        union { uint4 u[2]; v16bf v; } tb0, tb1, tb2, tb3;
        {
            const bf16* p0 = &Bsh[buf][wn * 4 + 0][lane][0];
            const bf16* p1 = &Bsh[buf][wn * 4 + 1][lane][0];
            const bf16* p2 = &Bsh[buf][wn * 4 + 2][lane][0];
            const bf16* p3 = &Bsh[buf][wn * 4 + 3][lane][0];
            tb0.u[0] = *(const uint4*)(p0); tb0.u[1] = *(const uint4*)(p0 + 8);
            tb1.u[0] = *(const uint4*)(p1); tb1.u[1] = *(const uint4*)(p1 + 8);
            tb2.u[0] = *(const uint4*)(p2); tb2.u[1] = *(const uint4*)(p2 + 8);
            tb3.u[0] = *(const uint4*)(p3); tb3.u[1] = *(const uint4*)(p3 + 8);
        }
        acc[0] = __builtin_amdgcn_wmma_f32_16x16x32_bf16(false, a, false, tb0.v, (short)0, acc[0], false, false);
        acc[1] = __builtin_amdgcn_wmma_f32_16x16x32_bf16(false, a, false, tb1.v, (short)0, acc[1], false, false);
        acc[2] = __builtin_amdgcn_wmma_f32_16x16x32_bf16(false, a, false, tb2.v, (short)0, acc[2], false, false);
        acc[3] = __builtin_amdgcn_wmma_f32_16x16x32_bf16(false, a, false, tb3.v, (short)0, acc[3], false, false);
        a = a_next;
        buf ^= 1;
    }

    const int row = m0 + ((lane >> 4) << 3);
#pragma unroll
    for (int i = 0; i < 4; ++i) {
        const int col = n0 + 16 * i + (lane & 15);
        const float bv = b_out[col];
#pragma unroll
        for (int r = 0; r < 8; ++r) {
            float v = sigmoidf_(acc[i][r] + bv);
            samples_t[(size_t)(row + r) * (T_ * C_) + col] = v;
            codes_bf[(size_t)(row + r) * C_ + col] = (bf16)v;
        }
    }
}

// ---------------------------------------------------------------------------
// Kernel: elementwise GRU cell update. Writes h (f32 + bf16) and hiddens slice.
// ---------------------------------------------------------------------------
__global__ __launch_bounds__(256) void gru_update_kernel(
    const float* __restrict__ gi, const float* __restrict__ gh,
    float* __restrict__ h_f32, bf16* __restrict__ h_bf,
    float* __restrict__ hiddens, int t)
{
    const int idx = blockIdx.x * blockDim.x + threadIdx.x;
    if (idx >= B_ * H_) return;
    const int b = idx / H_;
    const int j = idx - b * H_;
    const float* gib = gi + (size_t)b * G3H;
    const float* ghb = gh + (size_t)b * G3H;

    const float r  = sigmoidf_(gib[j]          + ghb[j]);
    const float z  = sigmoidf_(gib[H_ + j]     + ghb[H_ + j]);
    const float n  = tanhf    (gib[2 * H_ + j] + r * ghb[2 * H_ + j]);
    const float hp = h_f32[idx];
    const float hn = (1.0f - z) * n + z * hp;

    h_f32[idx] = hn;
    h_bf[idx]  = (bf16)hn;
    hiddens[(size_t)b * (T_ * H_) + (size_t)t * H_ + j] = hn;
}

// ---------------------------------------------------------------------------
// One-time helpers: weight transpose f32 [N,K] -> bf16 [K,N]; f32 -> bf16 copy.
// ---------------------------------------------------------------------------
__global__ __launch_bounds__(256) void transpose_to_bf16_kernel(
    const float* __restrict__ W, bf16* __restrict__ Wt, int N, int K)
{
    const int idx = blockIdx.x * blockDim.x + threadIdx.x;
    if (idx >= N * K) return;
    const int n = idx / K;
    const int k = idx - n * K;
    Wt[(size_t)k * N + n] = (bf16)W[idx];
}

__global__ __launch_bounds__(256) void cvt_f32_bf16_kernel(
    const float* __restrict__ src, bf16* __restrict__ dst, int n)
{
    const int idx = blockIdx.x * blockDim.x + threadIdx.x;
    if (idx < n) dst[idx] = (bf16)src[idx];
}

// ---------------------------------------------------------------------------
extern "C" void kernel_launch(void* const* d_in, const int* in_sizes, int n_in,
                              void* d_out, int out_size, void* d_ws, size_t ws_size,
                              hipStream_t stream)
{
    const float* noise = (const float*)d_in[0];   // [B, H]
    const float* W_ih  = (const float*)d_in[1];   // [3H, C]
    const float* b_ih  = (const float*)d_in[2];   // [3H]
    const float* W_hh  = (const float*)d_in[3];   // [3H, H]
    const float* b_hh  = (const float*)d_in[4];   // [3H]
    const float* W_out = (const float*)d_in[5];   // [C, H]
    const float* b_out = (const float*)d_in[6];   // [C]
    (void)in_sizes; (void)n_in; (void)out_size; (void)ws_size;

    float* samples = (float*)d_out;                         // [B, T, C]
    float* hiddens = samples + (size_t)B_ * T_ * C_;        // [B, T, H]

    char* ws = (char*)d_ws;
    size_t off = 0;
    auto alloc = [&](size_t bytes) -> void* {
        void* p = ws + off;
        off += (bytes + 255) & ~(size_t)255;
        return p;
    };
    bf16*  Wih_t    = (bf16*) alloc((size_t)C_ * G3H * sizeof(bf16));
    bf16*  Whh_t    = (bf16*) alloc((size_t)H_ * G3H * sizeof(bf16));
    bf16*  Wout_t   = (bf16*) alloc((size_t)H_ * C_  * sizeof(bf16));
    bf16*  codes_bf = (bf16*) alloc((size_t)B_ * C_  * sizeof(bf16));
    bf16*  h_bf     = (bf16*) alloc((size_t)B_ * H_  * sizeof(bf16));
    bf16*  noise_bf = (bf16*) alloc((size_t)B_ * H_  * sizeof(bf16));
    float* h_f32    = (float*)alloc((size_t)B_ * H_  * sizeof(float));
    float* gi       = (float*)alloc((size_t)B_ * G3H * sizeof(float));
    float* gh       = (float*)alloc((size_t)B_ * G3H * sizeof(float));

    // --- one-time setup: bf16 K-major weights, bf16 noise, h0 = 0 ---
    transpose_to_bf16_kernel<<<(G3H * C_ + 255) / 256, 256, 0, stream>>>(W_ih, Wih_t, G3H, C_);
    transpose_to_bf16_kernel<<<(G3H * H_ + 255) / 256, 256, 0, stream>>>(W_hh, Whh_t, G3H, H_);
    transpose_to_bf16_kernel<<<(C_  * H_ + 255) / 256, 256, 0, stream>>>(W_out, Wout_t, C_, H_);
    cvt_f32_bf16_kernel<<<(B_ * H_ + 255) / 256, 256, 0, stream>>>(noise, noise_bf, B_ * H_);
    hipMemsetAsync(h_f32, 0, (size_t)B_ * H_ * sizeof(float), stream);
    hipMemsetAsync(h_bf,  0, (size_t)B_ * H_ * sizeof(bf16),  stream);

    // codes0 = sigmoid(noise @ Wout^T + b_out) -> samples[:,0,:] and codes_bf
    const dim3 out_grid(C_ / 128, B_ / 64, 1);    // 16 x 4
    out_sigmoid_gemm_kernel<<<out_grid, 256, 0, stream>>>(
        noise_bf, Wout_t, b_out, samples, codes_bf);

    const dim3 gates_grid(G3H / 128, B_ / 64, 2); // 24 x 4 x 2
    for (int t = 0; t < T_; ++t) {
        gates_gemm_kernel<<<gates_grid, 256, 0, stream>>>(
            codes_bf, h_bf, Wih_t, Whh_t, b_ih, b_hh, gi, gh);
        gru_update_kernel<<<(B_ * H_ + 255) / 256, 256, 0, stream>>>(
            gi, gh, h_f32, h_bf, hiddens, t);
        if (t + 1 < T_) {
            out_sigmoid_gemm_kernel<<<out_grid, 256, 0, stream>>>(
                h_bf, Wout_t, b_out, samples + (size_t)(t + 1) * C_, codes_bf);
        }
    }
}
